// _AudioAttnBlock_27590869909718
// MI455X (gfx1250) — matrix operand order repaired and checked by reading, not compiled, not run on hardware
//
#include <hip/hip_runtime.h>

#define C     512
#define NPOS  4096
#define BATCH 4
#define GROUPS 32

typedef __bf16 bf16;
typedef bf16  v16bf __attribute__((ext_vector_type(16)));
typedef bf16  v8bf  __attribute__((ext_vector_type(8)));
typedef float v8f   __attribute__((ext_vector_type(8)));
typedef unsigned int v4u __attribute__((ext_vector_type(4)));
typedef int v8i __attribute__((ext_vector_type(8)));
typedef int v4i __attribute__((ext_vector_type(4)));

__device__ __forceinline__ v8f zero8() {
  v8f z;
#pragma unroll
  for (int i = 0; i < 8; ++i) z[i] = 0.f;
  return z;
}

__device__ __forceinline__ v16bf combine16(v8bf lo, v8bf hi8) {
  v16bf r;
#pragma unroll
  for (int i = 0; i < 8; ++i) { r[i] = lo[i]; r[i + 8] = hi8[i]; }
  return r;
}

// A-fragment (16x32 bf16): lane L row M=L&15; halves 0..7 = K hi*8.., 8..15 =
// K 16+hi*8.. (hi=L>>4). rowPtr -> (row_of_lane, k0), K-contiguous.
__device__ __forceinline__ v16bf load_a(const bf16* rowPtr, int hi) {
  v8bf x0 = *(const v8bf*)(rowPtr + hi * 8);
  v8bf x1 = *(const v8bf*)(rowPtr + 16 + hi * 8);
  return combine16(x0, x1);
}

// B-fragment (32x16 bf16): lane L col N=L&15; halves 0..7 = K hi*8.., 8..15 =
// K 16+hi*8.. (hi=L>>4). ptr -> (col_of_lane, k0), K-contiguous.
__device__ __forceinline__ v16bf load_b(const bf16* ptr, int hi) {
  v8bf y0 = *(const v8bf*)(ptr + hi * 8);
  v8bf y1 = *(const v8bf*)(ptr + 16 + hi * 8);
  return combine16(y0, y1);
}

__device__ __forceinline__ v8f wmma_bf16(v16bf a, v16bf b, v8f c) {
  return __builtin_amdgcn_wmma_f32_16x16x32_bf16(false, a, false, b, (short)0, c,
                                                 false, false);
}

// Async global->LDS copy (16B per lane), tracked by ASYNCcnt.
__device__ __forceinline__ void async_b128(void* ldsPtr, const void* gPtr) {
  unsigned l = (unsigned)(size_t)ldsPtr;          // low 32b = LDS offset
  unsigned long long g = (unsigned long long)(size_t)gPtr;
  asm volatile("global_load_async_to_lds_b128 %0, %1, off"
               :: "v"(l), "v"(g) : "memory");
}
__device__ __forceinline__ void wait_async0() {
  asm volatile("s_wait_asynccnt 0" ::: "memory");
}
__device__ __forceinline__ void wait_async8() {
  asm volatile("s_wait_asynccnt 8" ::: "memory");
}

// TDM: one 2D tile (256 dwords x 32 rows) global->LDS with hardware row
// padding (+4 dwords per 256) -> bank-conflict-free 520-bf16 row stride.
__device__ __forceinline__ void tdm_issue(const bf16* gsrc, unsigned ldsOff,
                                          v8i g1, v4i gz4, v8i gz8) {
  unsigned long long ga = (unsigned long long)(size_t)gsrc;
  v4u g0;
  g0[0] = 1u;                                         // count=1
  g0[1] = ldsOff;                                     // LDS byte address
  g0[2] = (unsigned)(ga & 0xFFFFFFFFu);
  g0[3] = ((unsigned)(ga >> 32) & 0x01FFFFFFu) | 0x80000000u;  // type=2
  __builtin_amdgcn_tensor_load_to_lds(g0, g1, gz4, gz4, gz8, 0);
}

// ---------------------------------------------------------------------------
// GroupNorm: x[B,C,H,W] f32 -> hT[B,N,C] bf16 (transposed, matmul-ready)
// ---------------------------------------------------------------------------
__global__ void gn_kernel(const float* __restrict__ x, const float* __restrict__ gamma,
                          const float* __restrict__ beta, bf16* __restrict__ hT) {
  const int g = blockIdx.x;
  const int b = blockIdx.y;
  const int t = threadIdx.x;  // 0..255
  const float* xg = x + ((size_t)b * C + g * 16) * NPOS;
  float s = 0.f, ss = 0.f;
#pragma unroll 4
  for (int i = t; i < 16 * NPOS; i += 256) { float v = xg[i]; s += v; ss += v * v; }
  __shared__ float red[2][8];
#pragma unroll
  for (int m = 16; m >= 1; m >>= 1) {
    s += __shfl_xor(s, m, 32);
    ss += __shfl_xor(ss, m, 32);
  }
  int lane = t & 31, wv = t >> 5;
  if (lane == 0) { red[0][wv] = s; red[1][wv] = ss; }
  __syncthreads();
  s = red[0][lane & 7]; ss = red[1][lane & 7];
#pragma unroll
  for (int m = 4; m >= 1; m >>= 1) {
    s += __shfl_xor(s, m, 32);
    ss += __shfl_xor(ss, m, 32);
  }
  const float invM = 1.f / (16 * NPOS);
  float mean = s * invM;
  float var = ss * invM - mean * mean;
  float rstd = rsqrtf(var + 1e-6f);
  int crel = t & 15;
  int c = g * 16 + crel;
  float ga = gamma[c] * rstd, be = beta[c] - mean * ga;
  const float* xc = xg + (size_t)crel * NPOS;
  bf16* hrow = hT + (size_t)b * NPOS * C + c;
  for (int p = t >> 4; p < NPOS; p += 16)
    hrow[(size_t)p * C] = (bf16)(xc[p] * ga + be);
}

// ---------------------------------------------------------------------------
// Weights f32 -> bf16 (Wq|Wk|Wv|Wo packed)
// ---------------------------------------------------------------------------
__global__ void wcvt_kernel(const float* __restrict__ Wq, const float* __restrict__ Wk,
                            const float* __restrict__ Wv, const float* __restrict__ Wo,
                            bf16* __restrict__ dst) {
  int idx = (blockIdx.x * 256 + threadIdx.x) * 4;
  int w = idx >> 18;
  int off = idx & 0x3FFFF;
  const float* src = (w == 0) ? Wq : (w == 1) ? Wk : (w == 2) ? Wv : Wo;
  float4 v = *(const float4*)(src + off);
  bf16* d = dst + (size_t)w * 262144 + off;
  d[0] = (bf16)v.x; d[1] = (bf16)v.y; d[2] = (bf16)v.z; d[3] = (bf16)v.w;
}

// ---------------------------------------------------------------------------
// q/k projection: outT[n,o] = sum_c hT[n,c]*W[o,c] + bias[o]; q scaled C^-0.5
// Double-buffered fragments so VMEM overlaps WMMA.
// ---------------------------------------------------------------------------
__global__ void proj_t_kernel(const bf16* __restrict__ hT, const bf16* __restrict__ Wq_bf,
                              const bf16* __restrict__ Wk_bf, const float* __restrict__ bq,
                              const float* __restrict__ bk, bf16* __restrict__ qT,
                              bf16* __restrict__ kT) {
  const int lane = threadIdx.x & 31, wv = threadIdx.x >> 5;
  const int hi = lane >> 4, lo = lane & 15;
  const int b = blockIdx.z >> 1, isK = blockIdx.z & 1;
  const bf16* W = isK ? Wk_bf : Wq_bf;
  const float* bias = isK ? bk : bq;
  bf16* out = (isK ? kT : qT) + (size_t)b * NPOS * C;
  const bf16* h = hT + (size_t)b * NPOS * C;
  const float scl = isK ? 1.f : rsqrtf((float)C);
  const int n0 = blockIdx.x * 64 + wv * 16;
  const int o0 = blockIdx.y * 64;
  v8f acc[4];
#pragma unroll
  for (int t = 0; t < 4; ++t) acc[t] = zero8();
  const bf16* arow = h + (size_t)(n0 + lo) * C;
  const bf16* brow = W + (size_t)(o0 + lo) * C;
  v16bf a_cur = load_a(arow, hi);
  v16bf b_cur[4];
#pragma unroll
  for (int t = 0; t < 4; ++t) b_cur[t] = load_b(brow + (size_t)t * 16 * C, hi);
  for (int ch = 0; ch < 16; ++ch) {
    const int nx = (ch + 1) & 15;
    v16bf a_nxt = load_a(arow + nx * 32, hi);
    v16bf b_nxt[4];
#pragma unroll
    for (int t = 0; t < 4; ++t)
      b_nxt[t] = load_b(brow + (size_t)t * 16 * C + nx * 32, hi);
#pragma unroll
    for (int t = 0; t < 4; ++t) acc[t] = wmma_bf16(a_cur, b_cur[t], acc[t]);
    a_cur = a_nxt;
#pragma unroll
    for (int t = 0; t < 4; ++t) b_cur[t] = b_nxt[t];
  }
#pragma unroll
  for (int t = 0; t < 4; ++t) {
    float bia = bias[o0 + t * 16 + lo];
#pragma unroll
    for (int r = 0; r < 8; ++r) {
      int n = n0 + r + hi * 8;
      out[(size_t)n * C + o0 + t * 16 + lo] = (bf16)((acc[t][r] + bia) * scl);
    }
  }
}

// ---------------------------------------------------------------------------
// v projection: v[o,n] = sum_c W[o,c]*hT[n,c] + bias[o] (kept [C,N])
// ---------------------------------------------------------------------------
__global__ void proj_v_kernel(const bf16* __restrict__ hT, const bf16* __restrict__ Wv_bf,
                              const float* __restrict__ bv, bf16* __restrict__ vO) {
  const int lane = threadIdx.x & 31, wv = threadIdx.x >> 5;
  const int hi = lane >> 4, lo = lane & 15;
  const int b = blockIdx.z;
  const int o0 = blockIdx.x * 64 + wv * 16;
  const int n0 = blockIdx.y * 64;
  const bf16* h = hT + (size_t)b * NPOS * C;
  bf16* out = vO + (size_t)b * C * NPOS;
  v8f acc[4];
#pragma unroll
  for (int t = 0; t < 4; ++t) acc[t] = zero8();
  const bf16* arow = Wv_bf + (size_t)(o0 + lo) * C;
  const bf16* brow = h + (size_t)(n0 + lo) * C;
  v16bf a_cur = load_a(arow, hi);
  v16bf b_cur[4];
#pragma unroll
  for (int t = 0; t < 4; ++t) b_cur[t] = load_b(brow + (size_t)t * 16 * C, hi);
  for (int ch = 0; ch < 16; ++ch) {
    const int nx = (ch + 1) & 15;
    v16bf a_nxt = load_a(arow + nx * 32, hi);
    v16bf b_nxt[4];
#pragma unroll
    for (int t = 0; t < 4; ++t)
      b_nxt[t] = load_b(brow + (size_t)t * 16 * C + nx * 32, hi);
#pragma unroll
    for (int t = 0; t < 4; ++t) acc[t] = wmma_bf16(a_cur, b_cur[t], acc[t]);
    a_cur = a_nxt;
#pragma unroll
    for (int t = 0; t < 4; ++t) b_cur[t] = b_nxt[t];
  }
#pragma unroll
  for (int t = 0; t < 4; ++t)
#pragma unroll
    for (int r = 0; r < 8; ++r) {
      int o = o0 + r + hi * 8;
      out[(size_t)o * NPOS + n0 + t * 16 + lo] = (bf16)(acc[t][r] + bv[o]);
    }
}

// ---------------------------------------------------------------------------
// S[i,j] = sum_c qT[i,c]*kT[j,c], then in-place softmax normalize -> P.
// k-tiles double-buffered in LDS via the Tensor Data Mover: tile i+1 streams
// while tile i is consumed; s_wait_tensorcnt(1) proves tile i landed (TDM ops
// from one wave complete in order).
// ---------------------------------------------------------------------------
#define LDK 520  // 512 + 8 bf16 pad = 1040B rows (260 dwords == 4 mod 64 banks)
#define KT_BUF (32 * LDK)
__global__ void attn_s_kernel(const bf16* __restrict__ qT, const bf16* __restrict__ kT,
                              bf16* __restrict__ S) {
  __shared__ bf16 lds[2 * KT_BUF];
  const int lane = threadIdx.x & 31, wv = threadIdx.x >> 5;
  const int hi = lane >> 4, lo = lane & 15;
  const int b = blockIdx.y;
  const int i0w = blockIdx.x * 64 + wv * 16;
  const bf16* q = qT + (size_t)b * NPOS * C;
  const bf16* k = kT + (size_t)b * NPOS * C;
  bf16* Sb = S + (size_t)b * NPOS * NPOS;

  // wave-resident q tile (16 rows x 512) = 128 VGPRs
  v16bf aq[16];
  const bf16* arow = q + (size_t)(i0w + lo) * C;
#pragma unroll
  for (int ch = 0; ch < 16; ++ch) aq[ch] = load_a(arow + ch * 32, hi);
  float rmax[8];
#pragma unroll
  for (int r = 0; r < 8; ++r) rmax[r] = -1e30f;

  // TDM descriptor group1: data_size=2 (4B units), pad_enable, pad_interval=7
  // (256 dw), pad_amount=3 (4 dw); tensor 256x4096, tile 256x32, stride0=256.
  v8i g1;
  g1[0] = 0x07D20000;
  g1[1] = 0x01000000;   // tensor_dim0=256 in bits[31:16]
  g1[2] = 0x10000000;   // tensor_dim1=4096 in bits[31:16]
  g1[3] = 0x01000000;   // tile_dim0=256 in bits[31:16]
  g1[4] = 32;           // tile_dim1 = 32 rows
  g1[5] = 256;          // tensor_dim0_stride = 256 (4B units)
  g1[6] = 0;
  g1[7] = 0;
  v4i gz4;
  gz4[0] = 0; gz4[1] = 0; gz4[2] = 0; gz4[3] = 0;
  v8i gz8;
#pragma unroll
  for (int i = 0; i < 8; ++i) gz8[i] = 0;
  const unsigned ldsBase = (unsigned)(size_t)(void*)lds;

  if (wv == 0) tdm_issue(k, ldsBase, g1, gz4, gz8);  // prologue: tile 0 -> buf 0

  for (int j0 = 0; j0 < NPOS; j0 += 32) {
    const int buf = (j0 >> 5) & 1;
    __syncthreads();  // prev compute done; buf^1 free to refill
    if (wv == 0) {
      if (j0 + 32 < NPOS) {
        tdm_issue(k + (size_t)(j0 + 32) * C, ldsBase + (buf ^ 1) * (KT_BUF * 2),
                  g1, gz4, gz8);
        __builtin_amdgcn_s_wait_tensorcnt(1);   // tile j0 complete
      } else {
        __builtin_amdgcn_s_wait_tensorcnt(0);
      }
    }
    __syncthreads();

    const bf16* kt = lds + buf * KT_BUF;
    v8f acc[2];
    acc[0] = zero8(); acc[1] = zero8();
    v16bf b_cur[2];
#pragma unroll
    for (int t = 0; t < 2; ++t)
      b_cur[t] = load_b(kt + (t * 16 + lo) * LDK, hi);
    for (int ch = 0; ch < 16; ++ch) {
      const int nx = (ch + 1) & 15;
      v16bf b_nxt[2];
#pragma unroll
      for (int t = 0; t < 2; ++t)
        b_nxt[t] = load_b(kt + (t * 16 + lo) * LDK + nx * 32, hi);
#pragma unroll
      for (int t = 0; t < 2; ++t) acc[t] = wmma_bf16(aq[ch], b_cur[t], acc[t]);
#pragma unroll
      for (int t = 0; t < 2; ++t) b_cur[t] = b_nxt[t];
    }
#pragma unroll
    for (int t = 0; t < 2; ++t)
#pragma unroll
      for (int r = 0; r < 8; ++r) {
        float sv = acc[t][r];
        rmax[r] = fmaxf(rmax[r], sv);
        Sb[(size_t)(i0w + r + hi * 8) * NPOS + j0 + t * 16 + lo] = (bf16)sv;
      }
  }
  // row-max reduction within each 16-lane half (rows r+8*hi)
#pragma unroll
  for (int r = 0; r < 8; ++r)
#pragma unroll
    for (int m = 8; m >= 1; m >>= 1)
      rmax[r] = fmaxf(rmax[r], __shfl_xor(rmax[r], m, 16));

  // softmax normalize: lane -> row = lane>>1 (same half as its rmax)
  int rowRel = lane >> 1, sub = lane & 1;
  float mr = rmax[0];
#pragma unroll
  for (int r = 1; r < 8; ++r) mr = ((rowRel & 7) == r) ? rmax[r] : mr;
  bf16* srow = Sb + (size_t)(i0w + rowRel) * NPOS + sub * 2048;
  float lsum = 0.f;
  for (int jj = 0; jj < 2048; jj += 8) {
    v8bf sv = *(const v8bf*)(srow + jj);
#pragma unroll
    for (int e = 0; e < 8; ++e) lsum += __expf((float)sv[e] - mr);
  }
  lsum += __shfl_xor(lsum, 1, 32);
  float rl = 1.f / lsum;
  for (int jj = 0; jj < 2048; jj += 8) {
    v8bf sv = *(const v8bf*)(srow + jj);
    v8bf pv;
#pragma unroll
    for (int e = 0; e < 8; ++e) pv[e] = (bf16)(__expf((float)sv[e] - mr) * rl);
    *(v8bf*)(srow + jj) = pv;
  }
}

// ---------------------------------------------------------------------------
// oT[i,c] = sum_j P[i,j]*v[c,j]. v j-blocks double-buffered with async-to-LDS
// loads: block i+1 streams (8 ops/wave in flight) while block i is consumed;
// s_wait_asynccnt 8 proves block i landed (async loads complete in order).
// ---------------------------------------------------------------------------
#define LDJ 136  // 128 + 8 bf16 pad
#define VT_BUF (64 * LDJ)
__global__ void attn_o_kernel(const bf16* __restrict__ P, const bf16* __restrict__ vO,
                              bf16* __restrict__ oT) {
  __shared__ bf16 lds[2 * VT_BUF];
  const int lane = threadIdx.x & 31, wv = threadIdx.x >> 5;
  const int hi = lane >> 4, lo = lane & 15;
  const int b = blockIdx.z;
  const int i0w = blockIdx.x * 64 + wv * 16;
  const int c0 = blockIdx.y * 64;
  const bf16* Pb = P + (size_t)b * NPOS * NPOS;
  const bf16* vb = vO + (size_t)b * C * NPOS;
  v8f acc[4];
#pragma unroll
  for (int t = 0; t < 4; ++t) acc[t] = zero8();
  const bf16* arow = Pb + (size_t)(i0w + lo) * NPOS;

  // prologue: stage block 0 into buf 0 (8 async ops per thread/wave)
#pragma unroll
  for (int idx = threadIdx.x; idx < 64 * 16; idx += 128) {
    int row = idx >> 4, c16 = idx & 15;
    async_b128(lds + row * LDJ + c16 * 8,
               vb + (size_t)(c0 + row) * NPOS + c16 * 8);
  }

  for (int jb = 0; jb < NPOS; jb += 128) {
    const int buf = (jb >> 7) & 1;
    __syncthreads();  // prev compute done; buf^1 free to refill
    if (jb + 128 < NPOS) {
      bf16* nbuf = lds + (buf ^ 1) * VT_BUF;
#pragma unroll
      for (int idx = threadIdx.x; idx < 64 * 16; idx += 128) {
        int row = idx >> 4, c16 = idx & 15;
        async_b128(nbuf + row * LDJ + c16 * 8,
                   vb + (size_t)(c0 + row) * NPOS + jb + 128 + c16 * 8);
      }
      wait_async8();   // 8 newer in flight; block jb's 8 complete
    } else {
      wait_async0();
    }
    __syncthreads();

    const bf16* vt = lds + buf * VT_BUF;
    v16bf b_cur[4];
#pragma unroll
    for (int t = 0; t < 4; ++t) b_cur[t] = load_b(vt + (t * 16 + lo) * LDJ, hi);
#pragma unroll
    for (int ch = 0; ch < 4; ++ch) {
      const int nx = (ch + 1) & 3;
      v16bf b_nxt[4];
#pragma unroll
      for (int t = 0; t < 4; ++t)
        b_nxt[t] = load_b(vt + (t * 16 + lo) * LDJ + nx * 32, hi);
      v16bf a = load_a(arow + jb + ch * 32, hi);
#pragma unroll
      for (int t = 0; t < 4; ++t) acc[t] = wmma_bf16(a, b_cur[t], acc[t]);
#pragma unroll
      for (int t = 0; t < 4; ++t) b_cur[t] = b_nxt[t];
    }
  }
#pragma unroll
  for (int t = 0; t < 4; ++t)
#pragma unroll
    for (int r = 0; r < 8; ++r)
      oT[(size_t)b * NPOS * C + (size_t)(i0w + r + hi * 8) * C + c0 + t * 16 + lo] =
          (bf16)acc[t][r];
}

// ---------------------------------------------------------------------------
// out[o,n] = x[o,n] + bo[o] + sum_c Wo[o,c]*oT[n,c]  (f32 output + residual)
// Whole 64x512 oT tile staged once via one async burst into a padded buffer.
// ---------------------------------------------------------------------------
__global__ void final_kernel(const bf16* __restrict__ oT, const bf16* __restrict__ Wo_bf,
                             const float* __restrict__ bo, const float* __restrict__ x,
                             float* __restrict__ out) {
  __shared__ bf16 lds[64 * LDK];
  const int lane = threadIdx.x & 31, wv = threadIdx.x >> 5;
  const int hi = lane >> 4, lo = lane & 15;
  const int b = blockIdx.z;
  const int o0w = blockIdx.x * 64 + wv * 16;
  const int n0 = blockIdx.y * 64;
  const bf16* ob = oT + (size_t)b * NPOS * C;
  v8f acc[4];
#pragma unroll
  for (int t = 0; t < 4; ++t) acc[t] = zero8();
  const bf16* arow = Wo_bf + (size_t)(o0w + lo) * C;

#pragma unroll
  for (int idx = threadIdx.x; idx < 64 * 64; idx += 128) {
    int row = idx >> 6, c16 = idx & 63;
    async_b128(lds + row * LDK + c16 * 8,
               ob + (size_t)(n0 + row) * C + c16 * 8);
  }
  wait_async0();
  __syncthreads();

  v16bf b_cur[4];
#pragma unroll
  for (int t = 0; t < 4; ++t) b_cur[t] = load_b(lds + (t * 16 + lo) * LDK, hi);
  for (int ch = 0; ch < 16; ++ch) {
    const int nx = (ch + 1) & 15;
    v16bf b_nxt[4];
#pragma unroll
    for (int t = 0; t < 4; ++t)
      b_nxt[t] = load_b(lds + (t * 16 + lo) * LDK + nx * 32, hi);
    v16bf a = load_a(arow + ch * 32, hi);
#pragma unroll
    for (int t = 0; t < 4; ++t) acc[t] = wmma_bf16(a, b_cur[t], acc[t]);
#pragma unroll
    for (int t = 0; t < 4; ++t) b_cur[t] = b_nxt[t];
  }

  const float* xb = x + (size_t)b * C * NPOS;
  float* outb = out + (size_t)b * C * NPOS;
#pragma unroll
  for (int t = 0; t < 4; ++t)
#pragma unroll
    for (int r = 0; r < 8; ++r) {
      int o = o0w + r + hi * 8;
      size_t off = (size_t)o * NPOS + n0 + t * 16 + lo;
      outb[off] = xb[off] + acc[t][r] + bo[o];
    }
}

extern "C" void kernel_launch(void* const* d_in, const int* in_sizes, int n_in,
                              void* d_out, int out_size, void* d_ws, size_t ws_size,
                              hipStream_t stream) {
  const float* x = (const float*)d_in[0];
  const float* gamma = (const float*)d_in[1];
  const float* beta = (const float*)d_in[2];
  const float* Wq = (const float*)d_in[3];
  const float* bq = (const float*)d_in[4];
  const float* Wk = (const float*)d_in[5];
  const float* bk = (const float*)d_in[6];
  const float* Wv = (const float*)d_in[7];
  const float* bv = (const float*)d_in[8];
  const float* Wo = (const float*)d_in[9];
  const float* bo = (const float*)d_in[10];
  float* out = (float*)d_out;

  char* ws = (char*)d_ws;
  bf16* wq_bf = (bf16*)ws;
  bf16* wk_bf = wq_bf + 262144;
  bf16* wv_bf = wq_bf + 2 * 262144;
  bf16* wo_bf = wq_bf + 3 * 262144;
  size_t off = (size_t)4 * 262144 * 2;
  const size_t actBytes = (size_t)BATCH * NPOS * C * 2;  // 16MB each
  bf16* hT = (bf16*)(ws + off); off += actBytes;
  bf16* qT = (bf16*)(ws + off); off += actBytes;
  bf16* kT = (bf16*)(ws + off); off += actBytes;
  bf16* vO = (bf16*)(ws + off); off += actBytes;
  bf16* oT = (bf16*)(ws + off); off += actBytes;
  bf16* S  = (bf16*)(ws + off);  // 128MB, lives in L2

  gn_kernel<<<dim3(GROUPS, BATCH), 256, 0, stream>>>(x, gamma, beta, hT);
  wcvt_kernel<<<1024, 256, 0, stream>>>(Wq, Wk, Wv, Wo, wq_bf);
  proj_t_kernel<<<dim3(64, 8, 8), 128, 0, stream>>>(hT, wq_bf, wk_bf, bq, bk, qT, kT);
  proj_v_kernel<<<dim3(8, 64, 4), 128, 0, stream>>>(hT, wv_bf, bv, vO);
  attn_s_kernel<<<dim3(64, 4), 128, 0, stream>>>(qT, kT, S);
  attn_o_kernel<<<dim3(64, 8, 4), 128, 0, stream>>>(S, vO, oT);
  final_kernel<<<dim3(8, 64, 4), 128, 0, stream>>>(oT, wo_bf, bo, x, out);
}
